// Distogram_3685081940005
// MI455X (gfx1250) — matrix operand order repaired and checked by reading, not compiled, not run on hardware
//
#include <hip/hip_runtime.h>
#include <hip/hip_bf16.h>

typedef __attribute__((ext_vector_type(16))) _Float16 v16h;
typedef __attribute__((ext_vector_type(8)))  float    v8f;

#define NLOC   1024
#define DIMD   256
#define DIMH   32
#define DIMS   64
#define CUTOFF 32

// ---------------- LDS layout (per block, main kernel) ----------------
// WHT : W_hidden^T  [32 cols][32 k] f16   = 2048 B   @ 0
// WOT : W_out^T     [64 cols][32 k] f16   = 4096 B   @ 2048
// LNS : ln_scale    32 f32                = 128 B    @ 6144
// LNB : ln_offset   32 f32                = 128 B    @ 6272
// per-wave region (8 waves), stride 3200:
//   pairTile  32 rows x 32 f16 (64B/row)  = 2048 B
//   hidTile   16 rows x 32 f16 (64B/row)  = 1024 B
//   mult      32 f32                      = 128 B
#define OFF_WHT   0
#define OFF_WOT   2048
#define OFF_LNS   6144
#define OFF_LNB   6272
#define OFF_WAVE  6400
#define WAVE_STRIDE 3200
#define OFF_PAIR  0
#define OFF_HID   2048
#define OFF_MULT  3072
#define SMEM_BYTES (OFF_WAVE + 8 * WAVE_STRIDE)   // 32000 B

__device__ __forceinline__ float gelu_f(float x) {
    // jax.nn.gelu default (approximate=True), tanh form
    float x3 = x * x * x;
    float t  = tanhf(0.7978845608028654f * (x + 0.044715f * x3));
    return 0.5f * x * (1.0f + t);
}

__device__ __forceinline__ unsigned int pack2h(float a, float b) {
    union { _Float16 h[2]; unsigned int u; } z;
    z.h[0] = (_Float16)a;
    z.h[1] = (_Float16)b;
    return z.u;
}

// B fragment (32x16 K x N, f16): lane L holds col = frag*16 + (L%16),
// K = (L/16)*16 .. +15 contiguous -> 32 bytes = 2 x uint4 from W^T LDS
// (W^T stored [col][k], 64 B per col row).
__device__ __forceinline__ v16h load_bfrag(const unsigned char* wt, int lane, int frag) {
    int col  = frag * 16 + (lane & 15);
    int byte = col * 64 + ((lane >> 4) * 32);
    v16h b;
    uint4* bp = reinterpret_cast<uint4*>(&b);
    bp[0] = *reinterpret_cast<const uint4*>(wt + byte);
    bp[1] = *reinterpret_cast<const uint4*>(wt + byte + 16);
    return b;
}

// A fragment (16x32 M x K, f16) from a row-major f16 tile with 64 B row
// stride: lane L holds row = row0 + (L%16); K = (L/16)*16 .. +15 contiguous.
__device__ __forceinline__ v16h load_afrag(const unsigned char* tile, int lane, int row0) {
    int row = row0 + (lane & 15);
    int kb  = (lane >> 4) * 32;          // bytes
    v16h a;
    uint4* ap = reinterpret_cast<uint4*>(&a);
    ap[0] = *reinterpret_cast<const uint4*>(tile + row * 64 + kb);
    ap[1] = *reinterpret_cast<const uint4*>(tile + row * 64 + kb + 16);
    return a;
}

// ---------------- Kernel 1: left/right projections into d_ws ----------------
__global__ void __launch_bounds__(64)
proj_kernel(const float* __restrict__ local,
            const float* __restrict__ Wl,
            const float* __restrict__ Wr,
            float* __restrict__ left,
            float* __restrict__ right) {
    __shared__ float lrow[DIMD];
    int n = blockIdx.x;
    for (int t = threadIdx.x; t < DIMD; t += 64)
        lrow[t] = local[n * DIMD + t];
    __syncthreads();
    int t = threadIdx.x;
    int h = t & 31;
    const float* W = (t < 32) ? Wl : Wr;
    float s = 0.0f;
#pragma unroll 8
    for (int k = 0; k < DIMD; ++k)
        s += lrow[k] * W[k * DIMH + h];
    if (t < 32) left[n * DIMH + h]  = s;
    else        right[n * DIMH + h] = s;
}

// ---------------- Kernel 2: pair MLP via WMMA ----------------
__global__ void __launch_bounds__(256)
pair_mlp_kernel(const float* __restrict__ leftp,
                const float* __restrict__ rightp,
                const int*   __restrict__ resi,
                const int*   __restrict__ chain,
                const int*   __restrict__ batch,
                const unsigned char* __restrict__ maskv,
                const float* __restrict__ W_relpos,
                const float* __restrict__ ln_scale,
                const float* __restrict__ ln_offset,
                const float* __restrict__ W_hidden,
                const float* __restrict__ W_out,
                float* __restrict__ outp) {
    __shared__ __align__(16) unsigned char smem[SMEM_BYTES];

    const int i    = blockIdx.x;
    const int lane = threadIdx.x & 31;
    const int wave = threadIdx.x >> 5;

    // ---- stage transposed f16 weights + LN params in LDS ----
    _Float16* wht = reinterpret_cast<_Float16*>(smem + OFF_WHT);
    _Float16* wot = reinterpret_cast<_Float16*>(smem + OFF_WOT);
    float*    lns = reinterpret_cast<float*>(smem + OFF_LNS);
    float*    lnb = reinterpret_cast<float*>(smem + OFF_LNB);
    for (int t = threadIdx.x; t < DIMH * DIMH; t += 256) {
        int k = t >> 5, n = t & 31;                 // W_hidden [k][n]
        wht[n * 32 + k] = (_Float16)W_hidden[k * DIMH + n];
    }
    for (int t = threadIdx.x; t < DIMH * DIMS; t += 256) {
        int k = t >> 6, n = t & 63;                 // W_out [k][n]
        wot[n * 32 + k] = (_Float16)W_out[k * DIMS + n];
    }
    if (threadIdx.x < DIMH) {
        lns[threadIdx.x] = ln_scale[threadIdx.x];
        lnb[threadIdx.x] = ln_offset[threadIdx.x];
    }
    __syncthreads();

    // loop-invariant B fragments in VGPRs
    v16h Bh0 = load_bfrag(smem + OFF_WHT, lane, 0);
    v16h Bh1 = load_bfrag(smem + OFF_WHT, lane, 1);
    v16h Bo0 = load_bfrag(smem + OFF_WOT, lane, 0);
    v16h Bo1 = load_bfrag(smem + OFF_WOT, lane, 1);
    v16h Bo2 = load_bfrag(smem + OFF_WOT, lane, 2);
    v16h Bo3 = load_bfrag(smem + OFF_WOT, lane, 3);

    unsigned char* wbase    = smem + OFF_WAVE + wave * WAVE_STRIDE;
    unsigned char* pairTile = wbase + OFF_PAIR;
    unsigned char* hidTile  = wbase + OFF_HID;
    float*         multArr  = reinterpret_cast<float*>(wbase + OFF_MULT);

    const int ri = resi[i];
    const int ci = chain[i];
    const int bi = batch[i];
    const int mi = (int)maskv[i];

    for (int it = 0; it < 4; ++it) {
        const int jbase = it * 256 + wave * 32;

        // ================= phase A: build LayerNorm'd pair rows =============
        {
            const int j  = jbase + lane;
            const int rj = resi[j];
            const int cj = chain[j];
            const int bj = batch[j];
            const int mj = (int)maskv[j];
            const bool sameb = (bi == bj);
            const bool samec = sameb && (ci == cj);
            int diff = ri - rj;
            diff = diff < -CUTOFF ? -CUTOFF : (diff > CUTOFF ? CUTOFF : diff);
            diff += CUTOFF;
            multArr[lane] = (sameb && mi && mj) ? 1.0f : 0.0f;

            const float relsc = samec ? 1.0f : 0.0f;
            const float4* lrow = reinterpret_cast<const float4*>(leftp + (size_t)i * DIMH);
            const float4* rrow = reinterpret_cast<const float4*>(rightp + (size_t)j * DIMH);
            const float4* relr = reinterpret_cast<const float4*>(W_relpos + (size_t)diff * DIMH);

            float acc[DIMH];
#pragma unroll
            for (int q = 0; q < 8; ++q) {
                float4 l = lrow[q], r = rrow[q], e = relr[q];
                acc[4 * q + 0] = l.x + r.x + relsc * e.x;
                acc[4 * q + 1] = l.y + r.y + relsc * e.y;
                acc[4 * q + 2] = l.z + r.z + relsc * e.z;
                acc[4 * q + 3] = l.w + r.w + relsc * e.w;
            }
            float mu = 0.0f;
#pragma unroll
            for (int h = 0; h < DIMH; ++h) mu += acc[h];
            mu *= (1.0f / 32.0f);
            float var = 0.0f;
#pragma unroll
            for (int h = 0; h < DIMH; ++h) { float d = acc[h] - mu; var += d * d; }
            var *= (1.0f / 32.0f);
            const float rs = rsqrtf(var + 1e-5f);

            uint4* prow = reinterpret_cast<uint4*>(pairTile + lane * 64);
#pragma unroll
            for (int q = 0; q < 4; ++q) {
                float y[8];
#pragma unroll
                for (int u = 0; u < 8; ++u) {
                    int h = 8 * q + u;
                    y[u] = (acc[h] - mu) * rs * lns[h] + lnb[h];
                }
                uint4 pv;
                pv.x = pack2h(y[0], y[1]);
                pv.y = pack2h(y[2], y[3]);
                pv.z = pack2h(y[4], y[5]);
                pv.w = pack2h(y[6], y[7]);
                prow[q] = pv;
            }
        }

        // ================= phase B: per-16-pair-group MLP ===================
        const int rlo = (lane >> 4) * 8;     // D-tile row offset for this lane
        const int c   = lane & 15;           // D-tile column for this lane
#pragma unroll
        for (int g = 0; g < 2; ++g) {
            v16h A = load_afrag(pairTile, lane, g * 16);
            v8f z = {};
            v8f D0 = __builtin_amdgcn_wmma_f32_16x16x32_f16(false, A, false, Bh0,
                                                            (short)0, z, false, false);
            v8f D1 = __builtin_amdgcn_wmma_f32_16x16x32_f16(false, A, false, Bh1,
                                                            (short)0, z, false, false);

            _Float16* hid = reinterpret_cast<_Float16*>(hidTile);
#pragma unroll
            for (int r = 0; r < 8; ++r) {
                int row16 = rlo + r;
                hid[row16 * 32 + c]      = (_Float16)gelu_f(D0[r]);
                hid[row16 * 32 + 16 + c] = (_Float16)gelu_f(D1[r]);
            }

            v16h Ah = load_afrag(hidTile, lane, 0);
            v8f E0 = __builtin_amdgcn_wmma_f32_16x16x32_f16(false, Ah, false, Bo0,
                                                            (short)0, z, false, false);
            v8f E1 = __builtin_amdgcn_wmma_f32_16x16x32_f16(false, Ah, false, Bo1,
                                                            (short)0, z, false, false);
            v8f E2 = __builtin_amdgcn_wmma_f32_16x16x32_f16(false, Ah, false, Bo2,
                                                            (short)0, z, false, false);
            v8f E3 = __builtin_amdgcn_wmma_f32_16x16x32_f16(false, Ah, false, Bo3,
                                                            (short)0, z, false, false);

#pragma unroll
            for (int r = 0; r < 8; ++r) {
                int row16 = rlo + r;
                float m = multArr[g * 16 + row16];
                long long j = (long long)jbase + g * 16 + row16;
                float* op = outp + ((long long)i * NLOC + j) * DIMS + c;
                op[0]  = E0[r] * m;
                op[16] = E1[r] * m;
                op[32] = E2[r] * m;
                op[48] = E3[r] * m;
            }
        }
    }
}

extern "C" void kernel_launch(void* const* d_in, const int* in_sizes, int n_in,
                              void* d_out, int out_size, void* d_ws, size_t ws_size,
                              hipStream_t stream) {
    const float* local   = (const float*)d_in[0];
    const int*   resi    = (const int*)d_in[1];
    const int*   chain   = (const int*)d_in[2];
    const int*   batch   = (const int*)d_in[3];
    const unsigned char* maskv = (const unsigned char*)d_in[4];
    const float* W_left  = (const float*)d_in[5];
    const float* W_right = (const float*)d_in[6];
    const float* W_relpos= (const float*)d_in[7];
    const float* ln_s    = (const float*)d_in[8];
    const float* ln_b    = (const float*)d_in[9];
    const float* W_hidden= (const float*)d_in[10];
    const float* W_out   = (const float*)d_in[11];

    float* left  = (float*)d_ws;                       // 1024*32 f32
    float* right = left + (size_t)NLOC * DIMH;         // 1024*32 f32

    proj_kernel<<<NLOC, 64, 0, stream>>>(local, W_left, W_right, left, right);

    pair_mlp_kernel<<<NLOC, 256, 0, stream>>>(left, right, resi, chain, batch,
                                              maskv, W_relpos, ln_s, ln_b,
                                              W_hidden, W_out, (float*)d_out);
    (void)in_sizes; (void)n_in; (void)out_size; (void)ws_size;
}
